// UniCrystalFormer_11888469476301
// MI455X (gfx1250) — matrix operand order repaired and checked by reading, not compiled, not run on hardware
//
#include <hip/hip_runtime.h>

// ---------------------------------------------------------------------------
// UniCrystalFormer forward for MI455X (gfx1250, wave32, WMMA).
// All dense linears run through v_wmma_f32_16x16x32_bf16 (bf16 operands,
// f32 accumulate). Norm stats / gates / scatter-adds stay in f32.
// GEMM is register-blocked 2x4 tiles per wave (32x64 output) for A/B reuse.
// ---------------------------------------------------------------------------

typedef __bf16 bf16_t;
typedef __attribute__((ext_vector_type(16))) __bf16 v16bf_t;
typedef __attribute__((ext_vector_type(8)))  __bf16 v8bf_t;
typedef __attribute__((ext_vector_type(8)))  float  v8f_t;

#define NNODES  8000
#define NEDGES  48000
#define NGRAPHS 64
#define HIDD    128
#define NHEADS  4
#define TCAT    384      // 3*C
#define QKV     512      // NH*C
#define NLAY    5
#define EPSV    1e-5f

#define ACT_NONE     0
#define ACT_SILU     1
#define ACT_SOFTPLUS 2
#define ACT_SIGMOID  3

// ---- small device helpers --------------------------------------------------

__device__ __forceinline__ bf16_t f2bf(float f) {
  union { float f; unsigned u; } x; x.f = f;
  unsigned r = x.u + 0x7FFFu + ((x.u >> 16) & 1u);   // round-to-nearest-even
  unsigned short h = (unsigned short)(r >> 16);
  return __builtin_bit_cast(bf16_t, h);
}

__device__ __forceinline__ float actf(float v, int act) {
  switch (act) {
    case ACT_SILU:     return v / (1.f + __expf(-v));
    case ACT_SOFTPLUS: return (v > 20.f) ? v : __logf(1.f + __expf(v));
    case ACT_SIGMOID:  return 1.f / (1.f + __expf(-v));
  }
  return v;
}

__device__ __forceinline__ float wave_sum(float v) {
#pragma unroll
  for (int m = 16; m >= 1; m >>= 1) v += __shfl_xor(v, m, 32);
  return v;
}

#define GS(i, n) for (int i = blockIdx.x * blockDim.x + threadIdx.x; i < (n); \
                      i += gridDim.x * blockDim.x)

// ---- register-blocked bf16 WMMA GEMM ---------------------------------------
// C[M,N] = act(A[M,K] @ W^T[N,K] + bias). Wave computes a 32x64 output block
// (2x4 sub-tiles of 16x16); K stepped by 32 with v_wmma_f32_16x16x32_bf16.
// Each A fragment feeds 4 WMMAs, each B fragment feeds 2 -> ~10.7 MAC/byte.
// A fragment layout (ISA 7.12.2, 16-bit A 16x32): lanes 0-15 row=l, K {0-7,16-23};
// lanes 16-31 row=l-16, K {8-15,24-31}. B fragment: lane%16 = column,
// same K split as A: lanes 0-15 K {0-7,16-23}, lanes 16-31 K {8-15,24-31}.
// Requires M%32==0, N%64==0, K%32==0 (true for every shape in this model).

__global__ __launch_bounds__(256) void k_gemm(
    const bf16_t* __restrict__ A, const bf16_t* __restrict__ WT,
    const float* __restrict__ bias, float* __restrict__ Cf,
    bf16_t* __restrict__ Cb, int M, int N, int K, int act)
{
  const int wave = blockIdx.x * 8 + (threadIdx.x >> 5);
  const int lane = threadIdx.x & 31;
  const int tiles_n = N >> 6;
  const int total   = (M >> 5) * tiles_n;
  if (wave >= total) return;                 // wave-uniform: EXEC stays all-ones
  const int tm   = wave / tiles_n;
  const int tn   = wave - tm * tiles_n;
  const int half = lane >> 4;
  const int l16  = lane & 15;

  const bf16_t* arow = A  + (size_t)(tm * 32 + l16) * K + half * 8;
  const bf16_t* bcol = WT + (size_t)(tn * 64 + l16) * K + half * 8;
  const size_t astr = (size_t)16 * K;   // next 16-row sub-tile
  const size_t bstr = (size_t)16 * K;   // next 16-col sub-tile

  v8f_t acc[2][4] = {};
  for (int k0 = 0; k0 < K; k0 += 32) {
    if (k0 + 32 < K) {
      __builtin_prefetch(arow + k0 + 32, 0, 1);          // global_prefetch_b8
      __builtin_prefetch(arow + astr + k0 + 32, 0, 1);
    }
    v16bf_t af[2], bfr[4];
#pragma unroll
    for (int i = 0; i < 2; ++i) {
      v8bf_t lo = *(const v8bf_t*)(arow + i * astr + k0);
      v8bf_t hi = *(const v8bf_t*)(arow + i * astr + k0 + 16);
#pragma unroll
      for (int t = 0; t < 8; ++t) { af[i][t] = lo[t]; af[i][t + 8] = hi[t]; }
    }
#pragma unroll
    for (int j = 0; j < 4; ++j) {
      v8bf_t lo = *(const v8bf_t*)(bcol + j * bstr + k0);
      v8bf_t hi = *(const v8bf_t*)(bcol + j * bstr + k0 + 16);
#pragma unroll
      for (int t = 0; t < 8; ++t) { bfr[j][t] = lo[t]; bfr[j][t + 8] = hi[t]; }
    }
#pragma unroll
    for (int i = 0; i < 2; ++i)
#pragma unroll
      for (int j = 0; j < 4; ++j)
        acc[i][j] = __builtin_amdgcn_wmma_f32_16x16x32_bf16(
            false, af[i], false, bfr[j], (short)0, acc[i][j], false, false);
  }

#pragma unroll
  for (int i = 0; i < 2; ++i) {
    const int row0 = tm * 32 + i * 16 + half * 8;  // lanes>=16 hold rows +8..+15
#pragma unroll
    for (int j = 0; j < 4; ++j) {
      const int col = tn * 64 + j * 16 + l16;
      const float bv = bias ? bias[col] : 0.f;
#pragma unroll
      for (int r = 0; r < 8; ++r) {
        float v = actf(acc[i][j][r] + bv, act);
        size_t off2 = (size_t)(row0 + r) * N + col;
        if (Cf) Cf[off2] = v;
        if (Cb) Cb[off2] = f2bf(v);
      }
    }
  }
}

// ---- weight prep: f32 [K,N] -> bf16 transposed [N,K] -----------------------
__global__ void k_w2bf(const float* __restrict__ w, bf16_t* __restrict__ wt,
                       int K, int N) {
  GS(i, K * N) {
    int n = i / K, k = i - n * K;
    wt[i] = f2bf(w[(size_t)k * N + n]);
  }
}

__global__ void k_zero(float* __restrict__ p, int n) { GS(i, n) p[i] = 0.f; }

__global__ void k_f2b(const float* __restrict__ in, bf16_t* __restrict__ out, int n) {
  GS(i, n) out[i] = f2bf(in[i]);
}

// ---- embedding gather ------------------------------------------------------
__global__ void k_embed(const int* __restrict__ xi, const float* __restrict__ emb,
                        float* __restrict__ xc, float* __restrict__ xm,
                        bf16_t* __restrict__ xcb, bf16_t* __restrict__ xmb) {
  GS(i, NNODES * HIDD) {
    int n = i >> 7, c = i & 127;
    float h = emb[(size_t)xi[n] * HIDD + c];
    xc[i] = h; xm[i] = h;
    bf16_t hb = f2bf(h);
    xcb[i] = hb; xmb[i] = hb;
  }
}

// ---- RBF expansion (wave per edge, 4 bins/lane) ----------------------------
__global__ __launch_bounds__(256) void k_rbf(const float* __restrict__ ea3,
                                             bf16_t* __restrict__ rbf) {
  int wave = blockIdx.x * 8 + (threadIdx.x >> 5);
  int lane = threadIdx.x & 31;
  if (wave >= NEDGES) return;
  float a = ea3[wave * 3], b = ea3[wave * 3 + 1], c = ea3[wave * 3 + 2];
  float d = sqrtf(a * a + b * b + c * c);
  const float step = 8.0f / 127.0f, gamma = 127.0f / 8.0f;
#pragma unroll
  for (int i = 0; i < 4; ++i) {
    int bin = lane + 32 * i;
    float t = d - step * bin;
    rbf[(size_t)wave * HIDD + bin] = f2bf(__expf(-gamma * t * t));
  }
}

// ---- CartNet concat: z = [x[dst], x[src], e] in bf16 -----------------------
__global__ void k_concat_cart(const bf16_t* __restrict__ xb,
                              const bf16_t* __restrict__ eb,
                              const int* __restrict__ src, const int* __restrict__ dst,
                              bf16_t* __restrict__ z) {
  GS(i, NEDGES * HIDD) {
    int e = i >> 7, c = i & 127;
    size_t zb = (size_t)e * TCAT;
    z[zb + c]            = xb[(size_t)dst[e] * HIDD + c];
    z[zb + HIDD + c]     = xb[(size_t)src[e] * HIDD + c];
    z[zb + 2 * HIDD + c] = eb[i];
  }
}

// ---- per-column batch stats (sum, sumsq) for BatchNorm ---------------------
__global__ __launch_bounds__(128) void k_colstats(const float* __restrict__ X,
                                                  int M, float* __restrict__ st) {
  int c = threadIdx.x;
  float s = 0.f, q = 0.f;
  for (int r = blockIdx.x; r < M; r += gridDim.x) {
    float v = X[(size_t)r * HIDD + c];
    s += v; q += v * v;
  }
  atomicAdd(&st[c], s);
  atomicAdd(&st[HIDD + c], q);
}

__device__ __forceinline__ float bn_apply(float v, const float* st, float minv,
                                          float g, float b, int c) {
  float mean = st[c] * minv;
  float var  = st[HIDD + c] * minv - mean * mean;
  return (v - mean) * rsqrtf(var + EPSV) * g + b;
}

// ---- cart: gate = sigmoid(BN(gpre)); ec += gate ----------------------------
__global__ void k_gate_update(const float* __restrict__ gpre, const float* __restrict__ st,
                              const float* __restrict__ g, const float* __restrict__ b,
                              float* __restrict__ gate, float* __restrict__ ec,
                              bf16_t* __restrict__ ecb) {
  const float minv = 1.f / NEDGES;
  GS(i, NEDGES * HIDD) {
    int c = i & 127;
    float y  = bn_apply(gpre[i], st, minv, g[c], b[c], c);
    float gt = 1.f / (1.f + __expf(-y));
    gate[i] = gt;
    float e2 = ec[i] + gt;
    ec[i] = e2; ecb[i] = f2bf(e2);
  }
}

// ---- cart: scatter msg = msgpre*gate into agg[dst] -------------------------
__global__ void k_msg_scatter(const float* __restrict__ msgpre,
                              const float* __restrict__ gate,
                              const int* __restrict__ dst, float* __restrict__ agg) {
  GS(i, NEDGES * HIDD) {
    int e = i >> 7, c = i & 127;
    atomicAdd(&agg[(size_t)dst[e] * HIDD + c], msgpre[i] * gate[i]);
  }
}

// ---- x_cart = silu(BN(agg)) + x_in -----------------------------------------
__global__ void k_bn_silu_res(const float* __restrict__ in, const float* __restrict__ st,
                              const float* __restrict__ g, const float* __restrict__ b,
                              const float* __restrict__ xin, float* __restrict__ xout,
                              bf16_t* __restrict__ xoutb) {
  const float minv = 1.f / NNODES;
  GS(i, NNODES * HIDD) {
    int c = i & 127;
    float y = bn_apply(in[i], st, minv, g[c], b[c], c);
    float s = y / (1.f + __expf(-y));
    float o = s + xin[i];
    xout[i] = o; xoutb[i] = f2bf(o);
  }
}

// ---- o = silu(BN(in)) (matformer node out) ---------------------------------
__global__ void k_bn_silu(const float* __restrict__ in, const float* __restrict__ st,
                          const float* __restrict__ g, const float* __restrict__ b,
                          float* __restrict__ out) {
  const float minv = 1.f / NNODES;
  GS(i, NNODES * HIDD) {
    int c = i & 127;
    float y = bn_apply(in[i], st, minv, g[c], b[c], c);
    out[i] = y / (1.f + __expf(-y));
  }
}

// ---- matformer: vc = [v[dst], v[src], ea] per (edge, head) in bf16 ---------
__global__ void k_vc_build(const float* __restrict__ v, const float* __restrict__ ea,
                           const int* __restrict__ src, const int* __restrict__ dst,
                           bf16_t* __restrict__ vc) {
  GS(i, NEDGES * NHEADS * HIDD) {
    int c = i & 127, h = (i >> 7) & 3, e = i >> 9;
    size_t row = (size_t)(e * NHEADS + h) * TCAT;
    size_t hoff = (size_t)h * HIDD + c;
    vc[row + c]            = f2bf(v[(size_t)dst[e] * QKV + hoff]);
    vc[row + HIDD + c]     = f2bf(v[(size_t)src[e] * QKV + hoff]);
    vc[row + 2 * HIDD + c] = f2bf(ea[(size_t)e * QKV + hoff]);
  }
}

// ---- matformer: alpha -> LN(384) -> sigmoid gate -> msgg = msgu * gate -----
__global__ __launch_bounds__(256) void k_alpha_gate(
    const float* __restrict__ q, const float* __restrict__ k,
    const float* __restrict__ ea, const int* __restrict__ src,
    const int* __restrict__ dst, const float* __restrict__ lng,
    const float* __restrict__ lnb, const float* __restrict__ msgu,
    bf16_t* __restrict__ msgg)
{
  int row  = blockIdx.x * 8 + (threadIdx.x >> 5);
  int lane = threadIdx.x & 31;
  if (row >= NEDGES * NHEADS) return;
  int e = row >> 2, h = row & 3;
  int d = dst[e], s = src[e];
  const float* qd = q  + (size_t)d * QKV + h * HIDD;
  const float* kd = k  + (size_t)d * QKV + h * HIDD;
  const float* ks = k  + (size_t)s * QKV + h * HIDD;
  const float* eb = ea + (size_t)e * QKV + h * HIDD;
  const float inv = 0.05103103630798287f;  // 1/sqrt(384)

  float a[12]; float sum = 0.f, sq = 0.f;
#pragma unroll
  for (int i = 0; i < 12; ++i) {
    int j = lane + 32 * i, c = j & 127, seg = j >> 7;
    float kv = (seg == 0) ? kd[c] : (seg == 1) ? ks[c] : eb[c];
    float av = qd[c] * kv * inv;
    a[i] = av; sum += av; sq += av * av;
  }
  sum = wave_sum(sum); sq = wave_sum(sq);
  float mean = sum * (1.f / 384.f);
  float var  = sq * (1.f / 384.f) - mean * mean;
  float rstd = rsqrtf(var + EPSV);
#pragma unroll
  for (int i = 0; i < 12; ++i) {
    int j = lane + 32 * i;
    float y = (a[i] - mean) * rstd * lng[j] + lnb[j];
    float g = 1.f / (1.f + __expf(-y));
    size_t off = (size_t)row * TCAT + j;
    msgg[off] = f2bf(msgu[off] * g);
  }
}

// ---- matformer: LN(128) on msg2 then scatter 0.25*msg into out[dst] --------
__global__ __launch_bounds__(256) void k_msgln_scatter(
    const float* __restrict__ msg2, const int* __restrict__ dst,
    const float* __restrict__ g, const float* __restrict__ b,
    float* __restrict__ out)
{
  int row  = blockIdx.x * 8 + (threadIdx.x >> 5);
  int lane = threadIdx.x & 31;
  if (row >= NEDGES * NHEADS) return;
  int e = row >> 2, d = dst[e];
  float a[4]; float sum = 0.f, sq = 0.f;
#pragma unroll
  for (int i = 0; i < 4; ++i) {
    int j = lane + 32 * i;
    float v = msg2[(size_t)row * HIDD + j];
    a[i] = v; sum += v; sq += v * v;
  }
  sum = wave_sum(sum); sq = wave_sum(sq);
  float mean = sum * (1.f / 128.f);
  float var  = sq * (1.f / 128.f) - mean * mean;
  float rstd = rsqrtf(var + EPSV);
#pragma unroll
  for (int i = 0; i < 4; ++i) {
    int j = lane + 32 * i;
    float y = (a[i] - mean) * rstd * g[j] + b[j];
    atomicAdd(&out[(size_t)d * HIDD + j], 0.25f * y);  // mean over 4 heads
  }
}

// ---- matformer tail: beta = sigmoid(bw . [o, xr, o-xr]); blend -------------
__global__ __launch_bounds__(256) void k_beta(
    const float* __restrict__ oN, const float* __restrict__ xrN,
    const float* __restrict__ bw, float* __restrict__ xmat,
    bf16_t* __restrict__ xmatb)
{
  int n    = blockIdx.x * 8 + (threadIdx.x >> 5);
  int lane = threadIdx.x & 31;
  if (n >= NNODES) return;
  const float* o  = oN  + (size_t)n * HIDD;
  const float* xr = xrN + (size_t)n * HIDD;
  float dotv = 0.f;
#pragma unroll
  for (int i = 0; i < 12; ++i) {
    int j = lane + 32 * i, c = j & 127, seg = j >> 7;
    float t = (seg == 0) ? o[c] : (seg == 1) ? xr[c] : (o[c] - xr[c]);
    dotv += t * bw[j];
  }
  dotv = wave_sum(dotv);
  float beta = 1.f / (1.f + __expf(-dotv));
#pragma unroll
  for (int i = 0; i < 4; ++i) {
    int c = lane + 32 * i;
    float xv = beta * xr[c] + (1.f - beta) * o[c];
    xmat[(size_t)n * HIDD + c]  = xv;
    xmatb[(size_t)n * HIDD + c] = f2bf(xv);
  }
}

// ---- mixer concat + update -------------------------------------------------
__global__ void k_concat2(const bf16_t* __restrict__ xcb, const bf16_t* __restrict__ xmb,
                          bf16_t* __restrict__ mixin) {
  GS(i, NNODES * HIDD) {
    int n = i >> 7, c = i & 127;
    mixin[(size_t)n * 256 + c]       = xcb[i];
    mixin[(size_t)n * 256 + 128 + c] = xmb[i];
  }
}

__global__ void k_mix_update(const float* __restrict__ g,
                             const float* __restrict__ xcart, const float* __restrict__ xmat,
                             float* __restrict__ xc, float* __restrict__ xm,
                             bf16_t* __restrict__ xcb, bf16_t* __restrict__ xmb) {
  GS(i, NNODES * HIDD) {
    float gv = g[i];
    float xo = gv * xcart[i] + (1.f - gv) * xmat[i];
    float cn = xo + xcart[i], mn = xo + xmat[i];
    xc[i] = cn; xm[i] = mn;
    xcb[i] = f2bf(cn); xmb[i] = f2bf(mn);
  }
}

// ---- pooling + head --------------------------------------------------------
__global__ void k_pool(const float* __restrict__ xc, const float* __restrict__ xm,
                       const int* __restrict__ batch, float* __restrict__ pooled,
                       float* __restrict__ counts) {
  GS(i, NNODES * HIDD) {
    int n = i >> 7, c = i & 127;
    float xf = 0.5f * (xc[i] + xm[i]);
    atomicAdd(&pooled[(size_t)batch[n] * HIDD + c], xf);
    if (c == 0) atomicAdd(&counts[batch[n]], 1.f);
  }
}

__global__ void k_pool_fin(const float* __restrict__ pooled, const float* __restrict__ counts,
                           bf16_t* __restrict__ poolb) {
  GS(i, NGRAPHS * HIDD) {
    int g = i >> 7;
    poolb[i] = f2bf(pooled[i] / fmaxf(counts[g], 1.f));
  }
}

__global__ __launch_bounds__(256) void k_fc2(const float* __restrict__ f1,
                                             const float* __restrict__ w2,
                                             const float* __restrict__ b2,
                                             float* __restrict__ out) {
  int g    = blockIdx.x * 8 + (threadIdx.x >> 5);
  int lane = threadIdx.x & 31;
  if (g >= NGRAPHS) return;
  float s = 0.f;
#pragma unroll
  for (int i = 0; i < 4; ++i) {
    int c = lane + 32 * i;
    s += f1[(size_t)g * HIDD + c] * w2[c];
  }
  s = wave_sum(s);
  if (lane == 0) out[g] = s + b2[0];
}

// ---------------------------------------------------------------------------
// host side
// ---------------------------------------------------------------------------

static inline int ewb(long n) { return (int)((n + 255) / 256); }

static void gemm(hipStream_t s, const bf16_t* A, const bf16_t* WT, const float* bias,
                 float* Cf, bf16_t* Cb, int M, int N, int K, int act) {
  int tiles = (M >> 5) * (N >> 6);          // 32x64 block per wave
  k_gemm<<<(tiles + 7) / 8, 256, 0, s>>>(A, WT, bias, Cf, Cb, M, N, K, act);
}

extern "C" void kernel_launch(void* const* d_in, const int* in_sizes, int n_in,
                              void* d_out, int out_size, void* d_ws, size_t ws_size,
                              hipStream_t stream)
{
  (void)in_sizes; (void)n_in; (void)out_size; (void)ws_size;

  // ---- input leaves (jax pytree flatten: sorted dict keys, lists in order)
  const int*   xi    = (const int*)d_in[0];
  const int*   eidx  = (const int*)d_in[1];
  const float* eattr = (const float*)d_in[2];
  const int*   batch = (const int*)d_in[3];
  const int*   src   = eidx;            // edge_index[0]
  const int*   dst   = eidx + NEDGES;   // edge_index[1]
  const float* emb   = (const float*)d_in[4];
  const float* fc1b  = (const float*)d_in[5];
  const float* fc1w  = (const float*)d_in[6];
  const float* fc2b  = (const float*)d_in[7];
  const float* fc2w  = (const float*)d_in[8];
  const float* rbf1b = (const float*)d_in[184];
  const float* rbf1w = (const float*)d_in[185];
  const float* rbf2b = (const float*)d_in[186];
  const float* rbf2w = (const float*)d_in[187];
  // per-layer leaf k within layer l (see mapping in analysis)
  auto Lp = [&](int l, int k) -> const float* {
    return (const float*)d_in[9 + 35 * l + k];
  };

  // ---- workspace arena -----------------------------------------------------
  char*  base = (char*)d_ws;
  size_t off  = 0;
  auto af = [&](size_t n) -> float*  { float*  p = (float*)(base + off);
                                       off += ((n * 4 + 255) & ~(size_t)255); return p; };
  auto ab = [&](size_t n) -> bf16_t* { bf16_t* p = (bf16_t*)(base + off);
                                       off += ((n * 2 + 255) & ~(size_t)255); return p; };

  float *xc    = af((size_t)NNODES * HIDD), *xm    = af((size_t)NNODES * HIDD);
  float *xcart = af((size_t)NNODES * HIDD), *xmat  = af((size_t)NNODES * HIDD);
  float *aggN  = af((size_t)NNODES * HIDD), *oN    = af((size_t)NNODES * HIDD);
  float *xrN   = af((size_t)NNODES * HIDD);
  float *ec    = af((size_t)NEDGES * HIDD);
  float *gateE = af((size_t)NEDGES * HIDD), *bufE  = af((size_t)NEDGES * HIDD);
  float *qf    = af((size_t)NNODES * QKV),  *kf    = af((size_t)NNODES * QKV);
  float *vf    = af((size_t)NNODES * QKV);
  float *eaf   = af((size_t)NEDGES * QKV);
  float *msgu  = af((size_t)NEDGES * NHEADS * TCAT);
  float *msg2  = af((size_t)NEDGES * NHEADS * HIDD);
  float *st    = af(2 * HIDD);
  float *pooled = af((size_t)NGRAPHS * HIDD), *counts = af(NGRAPHS);
  float *f1    = af((size_t)NGRAPHS * HIDD);

  bf16_t *xcb    = ab((size_t)NNODES * HIDD), *xmb    = ab((size_t)NNODES * HIDD);
  bf16_t *xcartb = ab((size_t)NNODES * HIDD), *xmatb  = ab((size_t)NNODES * HIDD);
  bf16_t *e0b    = ab((size_t)NEDGES * HIDD), *ecb    = ab((size_t)NEDGES * HIDD);
  bf16_t *rbfb   = ab((size_t)NEDGES * HIDD), *hEb    = ab((size_t)NEDGES * HIDD);
  bf16_t *zb     = ab((size_t)NEDGES * TCAT);
  bf16_t *vcb    = ab((size_t)NEDGES * NHEADS * TCAT);
  bf16_t *msggb  = ab((size_t)NEDGES * NHEADS * TCAT);
  bf16_t *mixinb = ab((size_t)NNODES * 2 * HIDD);
  bf16_t *poolb  = ab((size_t)NGRAPHS * HIDD);

  // per-layer transposed bf16 weights (reused across layers)
  bf16_t *Wg0 = ab((size_t)TCAT * HIDD), *Wg1 = ab((size_t)HIDD * HIDD);
  bf16_t *Wa0 = ab((size_t)TCAT * HIDD), *Wa1 = ab((size_t)HIDD * HIDD);
  bf16_t *Wq  = ab((size_t)HIDD * QKV),  *Wk  = ab((size_t)HIDD * QKV);
  bf16_t *Wv  = ab((size_t)HIDD * QKV),  *We  = ab((size_t)HIDD * QKV);
  bf16_t *Wupd = ab((size_t)TCAT * TCAT), *Wml = ab((size_t)TCAT * HIDD);
  bf16_t *Wsk = ab((size_t)HIDD * HIDD),  *Wmx = ab((size_t)2 * HIDD * HIDD);
  bf16_t *Wr1 = ab((size_t)HIDD * HIDD),  *Wr2 = ab((size_t)HIDD * HIDD);
  bf16_t *Wf1 = ab((size_t)HIDD * HIDD);

  auto tr = [&](const float* w, bf16_t* wt, int K, int N) {
    k_w2bf<<<ewb((long)K * N), 256, 0, stream>>>(w, wt, K, N);
  };

  // ---- embedding + RBF edge features --------------------------------------
  k_embed<<<ewb(NNODES * HIDD), 256, 0, stream>>>(xi, emb, xc, xm, xcb, xmb);
  k_rbf<<<(NEDGES + 7) / 8, 256, 0, stream>>>(eattr, rbfb);
  tr(rbf1w, Wr1, HIDD, HIDD);
  tr(rbf2w, Wr2, HIDD, HIDD);
  gemm(stream, rbfb, Wr1, rbf1b, nullptr, hEb, NEDGES, HIDD, HIDD, ACT_SOFTPLUS);
  gemm(stream, hEb,  Wr2, rbf2b, ec, e0b, NEDGES, HIDD, HIDD, ACT_NONE);
  k_f2b<<<ewb(NEDGES * HIDD), 256, 0, stream>>>(ec, ecb, NEDGES * HIDD);

  const int EH  = NEDGES * HIDD, NHd = NNODES * HIDD;
  const int ROWS = NEDGES * NHEADS;

  for (int l = 0; l < NLAY; ++l) {
    // weight prep (f32 -> bf16 [N,K])
    tr(Lp(l, 9),  Wg0, TCAT, HIDD);  tr(Lp(l, 11), Wg1, HIDD, HIDD);
    tr(Lp(l, 1),  Wa0, TCAT, HIDD);  tr(Lp(l, 3),  Wa1, HIDD, HIDD);
    tr(Lp(l, 28), Wq,  HIDD, QKV);   tr(Lp(l, 18), Wk,  HIDD, QKV);
    tr(Lp(l, 32), Wv,  HIDD, QKV);   tr(Lp(l, 16), We,  HIDD, QKV);
    tr(Lp(l, 26), Wupd, TCAT, TCAT); tr(Lp(l, 22), Wml, TCAT, HIDD);
    tr(Lp(l, 30), Wsk, HIDD, HIDD);  tr(Lp(l, 34), Wmx, 2 * HIDD, HIDD);

    // ---- CartNet branch ----
    k_concat_cart<<<ewb(EH), 256, 0, stream>>>(xcb, ecb, src, dst, zb);
    gemm(stream, zb,  Wg0, Lp(l, 8),  nullptr, hEb, NEDGES, HIDD, TCAT, ACT_SILU);
    gemm(stream, hEb, Wg1, Lp(l, 10), bufE, nullptr, NEDGES, HIDD, HIDD, ACT_NONE);
    k_zero<<<1, 256, 0, stream>>>(st, 2 * HIDD);
    k_colstats<<<128, 128, 0, stream>>>(bufE, NEDGES, st);
    k_gate_update<<<ewb(EH), 256, 0, stream>>>(bufE, st, Lp(l, 7), Lp(l, 6),
                                               gateE, ec, ecb);
    gemm(stream, zb,  Wa0, Lp(l, 0), nullptr, hEb, NEDGES, HIDD, TCAT, ACT_SILU);
    gemm(stream, hEb, Wa1, Lp(l, 2), bufE, nullptr, NEDGES, HIDD, HIDD, ACT_NONE);
    k_zero<<<ewb(NHd), 256, 0, stream>>>(aggN, NHd);
    k_msg_scatter<<<ewb(EH), 256, 0, stream>>>(bufE, gateE, dst, aggN);
    k_zero<<<1, 256, 0, stream>>>(st, 2 * HIDD);
    k_colstats<<<128, 128, 0, stream>>>(aggN, NNODES, st);
    k_bn_silu_res<<<ewb(NHd), 256, 0, stream>>>(aggN, st, Lp(l, 5), Lp(l, 4),
                                                xc, xcart, xcartb);

    // ---- Matformer branch ----
    gemm(stream, xmb, Wq, Lp(l, 27), qf, nullptr, NNODES, QKV, HIDD, ACT_NONE);
    gemm(stream, xmb, Wk, Lp(l, 17), kf, nullptr, NNODES, QKV, HIDD, ACT_NONE);
    gemm(stream, xmb, Wv, Lp(l, 31), vf, nullptr, NNODES, QKV, HIDD, ACT_NONE);
    gemm(stream, e0b, We, Lp(l, 15), eaf, nullptr, NEDGES, QKV, HIDD, ACT_NONE);
    k_vc_build<<<ewb((long)ROWS * HIDD), 256, 0, stream>>>(vf, eaf, src, dst, vcb);
    gemm(stream, vcb, Wupd, Lp(l, 25), msgu, nullptr, ROWS, TCAT, TCAT, ACT_NONE);
    k_alpha_gate<<<(ROWS + 7) / 8, 256, 0, stream>>>(qf, kf, eaf, src, dst,
                                                     Lp(l, 20), Lp(l, 19), msgu, msggb);
    gemm(stream, msggb, Wml, Lp(l, 21), msg2, nullptr, ROWS, HIDD, TCAT, ACT_NONE);
    k_zero<<<ewb(NHd), 256, 0, stream>>>(aggN, NHd);
    k_msgln_scatter<<<(ROWS + 7) / 8, 256, 0, stream>>>(msg2, dst, Lp(l, 24),
                                                        Lp(l, 23), aggN);
    k_zero<<<1, 256, 0, stream>>>(st, 2 * HIDD);
    k_colstats<<<128, 128, 0, stream>>>(aggN, NNODES, st);
    k_bn_silu<<<ewb(NHd), 256, 0, stream>>>(aggN, st, Lp(l, 14), Lp(l, 13), oN);
    gemm(stream, xmb, Wsk, Lp(l, 29), xrN, nullptr, NNODES, HIDD, HIDD, ACT_NONE);
    k_beta<<<(NNODES + 7) / 8, 256, 0, stream>>>(oN, xrN, Lp(l, 12), xmat, xmatb);

    // ---- residual gate mixer ----
    k_concat2<<<ewb(NHd), 256, 0, stream>>>(xcartb, xmatb, mixinb);
    gemm(stream, mixinb, Wmx, Lp(l, 33), oN, nullptr, NNODES, HIDD, 2 * HIDD,
         ACT_SIGMOID);
    k_mix_update<<<ewb(NHd), 256, 0, stream>>>(oN, xcart, xmat, xc, xm, xcb, xmb);
  }

  // ---- readout -------------------------------------------------------------
  k_zero<<<ewb(NGRAPHS * HIDD), 256, 0, stream>>>(pooled, NGRAPHS * HIDD);
  k_zero<<<1, 64, 0, stream>>>(counts, NGRAPHS);
  k_pool<<<ewb(NHd), 256, 0, stream>>>(xc, xm, batch, pooled, counts);
  k_pool_fin<<<ewb(NGRAPHS * HIDD), 256, 0, stream>>>(pooled, counts, poolb);
  tr(fc1w, Wf1, HIDD, HIDD);
  gemm(stream, poolb, Wf1, fc1b, f1, nullptr, NGRAPHS, HIDD, HIDD, ACT_SILU);
  k_fc2<<<(NGRAPHS + 7) / 8, 256, 0, stream>>>(f1, fc2w, fc2b, (float*)d_out);
}
